// MSA_26328149524927
// MI455X (gfx1250) — matrix operand order repaired and checked by reading, not compiled, not run on hardware
//
#include <hip/hip_runtime.h>

#define EMBED 1024
#define NH 16
#define HD 64
#define SEQ 2048
#define BATCH 4
#define ROWS (BATCH * SEQ) /* 8192 */

typedef __attribute__((ext_vector_type(16))) _Float16 v16h;
typedef __attribute__((ext_vector_type(8)))  _Float16 v8h;
typedef __attribute__((ext_vector_type(8)))  float    v8f;

union AF { v16h v; v8h h[2]; };

// ---- WMMA fragment loaders (wave32, 16x16x32 f16 shapes) -------------------
// A (16x32, MxK) row-major, ld = lda halves.
// Layout: lane m = lane&15; elem e -> k = (e/8)*16 + (lane/16)*8 + (e%8).
__device__ inline v16h load_a_frag(const _Float16* base, int lda, int lane) {
  int m  = lane & 15;
  int kq = (lane >> 4) * 8;
  AF f;
  f.h[0] = *(const v8h*)(base + m * lda + kq);
  f.h[1] = *(const v8h*)(base + m * lda + 16 + kq);
  return f.v;
}

// B (32x16, KxN) stored N-major: elem(k,n) = base[n*ldb + k].
// Layout: lane n = lane&15; elem e -> k = (e/8)*16 + (lane/16)*8 + (e%8).
__device__ inline v16h load_bt_frag(const _Float16* base, int ldb, int lane) {
  int n  = lane & 15;
  int k0 = (lane >> 4) * 8;
  AF f;
  f.h[0] = *(const v8h*)(base + n * ldb + k0);
  f.h[1] = *(const v8h*)(base + n * ldb + k0 + 16);
  return f.v;
}

__device__ inline v8f wmma16(v16h a, v16h b, v8f c) {
  return __builtin_amdgcn_wmma_f32_16x16x32_f16(false, a, false, b, (short)0, c,
                                                false, false);
}

// Async copy of one 16-byte chunk: global -> LDS, tracked by ASYNCcnt.
__device__ inline void async_b128(unsigned lds_addr, const void* gaddr) {
  asm volatile("global_load_async_to_lds_b128 %0, %1, off"
               :: "v"(lds_addr), "v"(gaddr) : "memory");
}
__device__ inline void wait_async0() {
  asm volatile("s_wait_asynccnt 0x0" ::: "memory");
}
__device__ inline unsigned lds_off(const void* p) {
  return (unsigned)(size_t)p; // low 32 bits of a shared generic address = LDS offset
}

// ---- conversion kernels ----------------------------------------------------
__global__ void f32_to_f16_k(const float* __restrict__ in,
                             _Float16* __restrict__ out, int n) {
  for (int i = blockIdx.x * blockDim.x + threadIdx.x; i < n;
       i += gridDim.x * blockDim.x)
    out[i] = (_Float16)in[i];
}

// Wt[n*1024 + k] = (f16) W[k*1024 + n]  (exactly 1M threads)
__global__ void transpose_f16_k(const float* __restrict__ in,
                                _Float16* __restrict__ out) {
  int i = blockIdx.x * blockDim.x + threadIdx.x;
  int n = i >> 10, k = i & 1023;
  out[(long)n * EMBED + k] = (_Float16)in[(long)k * EMBED + n];
}

// ---- QKV projection: C[8192,1024] = xh @ W + b, scatter to attention layout
// Block = 128 rows x 64 cols; wave = 32 rows x 64 cols.
// Weight panel (64 n x 32 k, N-major) async-staged to LDS, double-buffered.
// mode 0/1: out[((b*NH+h)*SEQ+s)*HD + d]   (Q, K)
// mode 2  : out[((b*NH+h)*HD+d)*SEQ + s]   (V transposed for PV B-fragments)
__global__ __launch_bounds__(128) void gemm_qkv_k(
    const _Float16* __restrict__ xh, const _Float16* __restrict__ wt,
    const float* __restrict__ bias, _Float16* __restrict__ out, int mode) {
  __shared__ __align__(16) _Float16 Bs[2][64 * 32]; // [n][k], ld = 32 halves
  int lane = threadIdx.x & 31;
  int wid  = threadIdx.x >> 5;
  int nt  = blockIdx.x & 15;   // 16 col-panels of 64
  int mtb = blockIdx.x >> 4;   // 64 row-panels of 128
  int n0 = nt * 64;
  int m0 = mtb * 128 + wid * 32;

  auto issueB = [&](int k, int sel) {
    const char* src = (const char*)(wt + (long)n0 * EMBED + k);
    unsigned dl = lds_off(&Bs[sel][0]);
#pragma unroll
    for (int i = 0; i < 2; ++i) {
      int ti = (wid * 2 + i) * 32 + lane; // 0..255: row=n, 4 chunks of 16B
      int row = ti >> 2, c = ti & 3;
      async_b128(dl + row * 64 + c * 16,
                 src + (long)row * EMBED * 2 + c * 16);
    }
  };

  v8f acc[2][4] = {};
  issueB(0, 0);
  wait_async0();
  __syncthreads();
  for (int k = 0; k < EMBED; k += 32) {
    int sel = (k >> 5) & 1;
    if (k + 32 < EMBED) issueB(k + 32, sel ^ 1);
    v16h a0 = load_a_frag(xh + (long)m0 * EMBED + k, EMBED, lane);
    v16h a1 = load_a_frag(xh + (long)(m0 + 16) * EMBED + k, EMBED, lane);
#pragma unroll
    for (int j = 0; j < 4; ++j) {
      v16h bb = load_bt_frag(&Bs[sel][(j * 16) * 32], 32, lane);
      acc[0][j] = wmma16(a0, bb, acc[0][j]);
      acc[1][j] = wmma16(a1, bb, acc[1][j]);
    }
    wait_async0();
    __syncthreads();
  }

  int half = lane >> 4, nl = lane & 15;
#pragma unroll
  for (int sub = 0; sub < 2; ++sub) {
    int mb = m0 + sub * 16;
#pragma unroll
    for (int j = 0; j < 4; ++j) {
      int n = n0 + j * 16 + nl;
      float bv = bias[n];
      int h = n >> 6, d = n & 63;
#pragma unroll
      for (int r = 0; r < 8; ++r) {
        int m = mb + r + half * 8;
        int bi = m >> 11; // m / SEQ
        int s  = m & 2047;
        float val = acc[sub][j][r] + bv;
        long idx = (mode == 2)
                       ? ((long)(bi * NH + h) * HD + d) * SEQ + s
                       : ((long)(bi * NH + h) * SEQ + s) * HD + d;
        out[idx] = (_Float16)val;
      }
    }
  }
}

// ---- flash attention: block = 64 query rows of one (b,h); wave = 16 rows ---
// K chunk (32t x 64d, contiguous) and V chunk (64d x 32t from Vt) are
// async-staged to double-buffered LDS, shared by all 4 waves.
__global__ __launch_bounds__(128) void attn_k(const _Float16* __restrict__ Q,
                                              const _Float16* __restrict__ K,
                                              const _Float16* __restrict__ Vt,
                                              _Float16* __restrict__ O) {
  __shared__ __align__(16) _Float16 Ks[2][32 * 64];   // [t][d], ld = 64
  __shared__ __align__(16) _Float16 Vs[2][64 * 32];   // [d][t], ld = 32
  __shared__ __align__(16) _Float16 Ptile[4][16 * 32];
  int lane = threadIdx.x & 31;
  int wid  = threadIdx.x >> 5;
  int wave = blockIdx.x * 4 + wid; // 0..8191; block shares (b,h)
  int stile = wave & 127;
  int h = (wave >> 7) & 15;
  int b = wave >> 11;

  const _Float16* Qb = Q  + (long)(b * NH + h) * SEQ * HD;
  const _Float16* Kb = K  + (long)(b * NH + h) * SEQ * HD;
  const _Float16* Vb = Vt + (long)(b * NH + h) * HD * SEQ;
  _Float16* pl = Ptile[wid];
  int s0 = stile * 16;
  int half = lane >> 4, nl = lane & 15;

  auto issueKV = [&](int t, int sel) {
    // K rows t..t+31, all 64 d: 4KB contiguous
    const char* ksrc = (const char*)(Kb + (long)t * HD);
    unsigned kl = lds_off(&Ks[sel][0]);
#pragma unroll
    for (int i = 0; i < 2; ++i) {
      unsigned fb = (unsigned)(((wid * 2 + i) * 32 + lane) * 16);
      async_b128(kl + fb, ksrc + fb);
    }
    // V rows d=0..63, cols t..t+31 (row stride SEQ halves)
    const char* vsrc = (const char*)Vb + (long)t * 2;
    unsigned vl = lds_off(&Vs[sel][0]);
#pragma unroll
    for (int i = 0; i < 2; ++i) {
      int ti = (wid * 2 + i) * 32 + lane; // 0..255: row=d, 4 chunks of 16B
      int row = ti >> 2, c = ti & 3;
      async_b128(vl + row * 64 + c * 16,
                 vsrc + (long)row * SEQ * 2 + c * 16);
    }
  };

  v16h qa0 = load_a_frag(Qb + (long)s0 * HD + 0, HD, lane);
  v16h qa1 = load_a_frag(Qb + (long)s0 * HD + 32, HD, lane);

  v8f o[4] = {};
  float mrun[8], lrun[8];
#pragma unroll
  for (int r = 0; r < 8; ++r) { mrun[r] = -1e30f; lrun[r] = 0.f; }
  const float scale = 0.125f; // 1/sqrt(64)

  issueKV(0, 0);
  wait_async0();
  __syncthreads();

  for (int t = 0; t < SEQ; t += 32) {
    int sel = (t >> 5) & 1;
    if (t + 32 < SEQ) issueKV(t + 32, sel ^ 1);

    // scores: two 16x16 C tiles over t..t+31, K-dim 64 in 2 steps (from LDS)
    v8f sA = {}, sB = {};
    {
      v16h kb = load_bt_frag(&Ks[sel][0 * 64 + 0], 64, lane);
      sA = wmma16(qa0, kb, sA);
      kb = load_bt_frag(&Ks[sel][0 * 64 + 32], 64, lane);
      sA = wmma16(qa1, kb, sA);
      kb = load_bt_frag(&Ks[sel][16 * 64 + 0], 64, lane);
      sB = wmma16(qa0, kb, sB);
      kb = load_bt_frag(&Ks[sel][16 * 64 + 32], 64, lane);
      sB = wmma16(qa1, kb, sB);
    }
    float f0[8], f1[8], mnew[8], alpha[8], e0[8], e1[8];
#pragma unroll
    for (int r = 0; r < 8; ++r) { f0[r] = sA[r] * scale; f1[r] = sB[r] * scale; }
#pragma unroll
    for (int r = 0; r < 8; ++r) {
      float mx = fmaxf(f0[r], f1[r]);
#pragma unroll
      for (int off = 8; off >= 1; off >>= 1)
        mx = fmaxf(mx, __shfl_xor(mx, off, 32));
      mnew[r]  = fmaxf(mrun[r], mx);
      alpha[r] = __expf(mrun[r] - mnew[r]);
      mrun[r]  = mnew[r];
      e0[r] = __expf(f0[r] - mnew[r]);
      e1[r] = __expf(f1[r] - mnew[r]);
      float sum = e0[r] + e1[r];
#pragma unroll
      for (int off = 8; off >= 1; off >>= 1)
        sum += __shfl_xor(sum, off, 32);
      lrun[r] = lrun[r] * alpha[r] + sum;
    }
#pragma unroll
    for (int j = 0; j < 4; ++j)
#pragma unroll
      for (int r = 0; r < 8; ++r) o[j][r] *= alpha[r];

    // P (16x32 f16) -> LDS row-major, then re-read as an A fragment
#pragma unroll
    for (int r = 0; r < 8; ++r) {
      int m = r + half * 8;
      pl[m * 32 + nl]      = (_Float16)e0[r];
      pl[m * 32 + 16 + nl] = (_Float16)e1[r];
    }
    asm volatile("s_wait_dscnt 0x0" ::: "memory");
    v16h pa = load_a_frag(pl, 32, lane);
#pragma unroll
    for (int j = 0; j < 4; ++j) {
      v16h vb = load_bt_frag(&Vs[sel][(j * 16) * 32], 32, lane);
      o[j] = wmma16(pa, vb, o[j]);
    }

    wait_async0();
    __syncthreads();
  }
  // normalize and store O as f16 [B,S,E]
#pragma unroll
  for (int j = 0; j < 4; ++j) {
#pragma unroll
    for (int r = 0; r < 8; ++r) {
      int s = s0 + r + half * 8;
      float val = o[j][r] / lrun[r];
      O[((long)(b * SEQ + s)) * EMBED + h * HD + j * 16 + nl] = (_Float16)val;
    }
  }
}

// ---- output projection: fp32 out = O_h @ Wo + bo (same LDS-staged scheme) --
__global__ __launch_bounds__(128) void gemm_oproj_k(
    const _Float16* __restrict__ oh, const _Float16* __restrict__ wt,
    const float* __restrict__ bias, float* __restrict__ out) {
  __shared__ __align__(16) _Float16 Bs[2][64 * 32];
  int lane = threadIdx.x & 31;
  int wid  = threadIdx.x >> 5;
  int nt  = blockIdx.x & 15;
  int mtb = blockIdx.x >> 4;
  int n0 = nt * 64;
  int m0 = mtb * 128 + wid * 32;

  auto issueB = [&](int k, int sel) {
    const char* src = (const char*)(wt + (long)n0 * EMBED + k);
    unsigned dl = lds_off(&Bs[sel][0]);
#pragma unroll
    for (int i = 0; i < 2; ++i) {
      int ti = (wid * 2 + i) * 32 + lane;
      int row = ti >> 2, c = ti & 3;
      async_b128(dl + row * 64 + c * 16,
                 src + (long)row * EMBED * 2 + c * 16);
    }
  };

  v8f acc[2][4] = {};
  issueB(0, 0);
  wait_async0();
  __syncthreads();
  for (int k = 0; k < EMBED; k += 32) {
    int sel = (k >> 5) & 1;
    if (k + 32 < EMBED) issueB(k + 32, sel ^ 1);
    v16h a0 = load_a_frag(oh + (long)m0 * EMBED + k, EMBED, lane);
    v16h a1 = load_a_frag(oh + (long)(m0 + 16) * EMBED + k, EMBED, lane);
#pragma unroll
    for (int j = 0; j < 4; ++j) {
      v16h bb = load_bt_frag(&Bs[sel][(j * 16) * 32], 32, lane);
      acc[0][j] = wmma16(a0, bb, acc[0][j]);
      acc[1][j] = wmma16(a1, bb, acc[1][j]);
    }
    wait_async0();
    __syncthreads();
  }

  int half = lane >> 4, nl = lane & 15;
#pragma unroll
  for (int sub = 0; sub < 2; ++sub) {
    int mb = m0 + sub * 16;
#pragma unroll
    for (int j = 0; j < 4; ++j) {
      int n = n0 + j * 16 + nl;
      float bv = bias[n];
#pragma unroll
      for (int r = 0; r < 8; ++r) {
        int m = mb + r + half * 8;
        out[(long)m * EMBED + n] = acc[sub][j][r] + bv;
      }
    }
  }
}

extern "C" void kernel_launch(void* const* d_in, const int* in_sizes, int n_in,
                              void* d_out, int out_size, void* d_ws,
                              size_t ws_size, hipStream_t stream) {
  const float* x  = (const float*)d_in[0];
  const float* Wq = (const float*)d_in[1];
  const float* bq = (const float*)d_in[2];
  const float* Wk = (const float*)d_in[3];
  const float* bk = (const float*)d_in[4];
  const float* Wv = (const float*)d_in[5];
  const float* bv = (const float*)d_in[6];
  const float* Wo = (const float*)d_in[7];
  const float* bo = (const float*)d_in[8];

  char* ws = (char*)d_ws;
  const size_t MB = 1024 * 1024;
  _Float16* xh  = (_Float16*)(ws + 0);        // 16 MB
  _Float16* wqt = (_Float16*)(ws + 16 * MB);  // 2 MB each
  _Float16* wkt = (_Float16*)(ws + 18 * MB);
  _Float16* wvt = (_Float16*)(ws + 20 * MB);
  _Float16* wot = (_Float16*)(ws + 22 * MB);
  _Float16* Qb  = (_Float16*)(ws + 24 * MB);  // 16 MB
  _Float16* Kb  = (_Float16*)(ws + 40 * MB);  // 16 MB
  _Float16* Vtb = (_Float16*)(ws + 56 * MB);  // 16 MB (transposed [B,H,D,S])
  _Float16* Ob  = (_Float16*)(ws + 72 * MB);  // 16 MB

  f32_to_f16_k<<<4096, 256, 0, stream>>>(x, xh, ROWS * EMBED);
  transpose_f16_k<<<4096, 256, 0, stream>>>(Wq, wqt);
  transpose_f16_k<<<4096, 256, 0, stream>>>(Wk, wkt);
  transpose_f16_k<<<4096, 256, 0, stream>>>(Wv, wvt);
  transpose_f16_k<<<4096, 256, 0, stream>>>(Wo, wot);

  gemm_qkv_k<<<1024, 128, 0, stream>>>(xh, wqt, bq, Qb, 0);
  gemm_qkv_k<<<1024, 128, 0, stream>>>(xh, wkt, bk, Kb, 1);
  gemm_qkv_k<<<1024, 128, 0, stream>>>(xh, wvt, bv, Vtb, 2);

  attn_k<<<2048, 128, 0, stream>>>(Qb, Kb, Vtb, Ob);

  gemm_oproj_k<<<1024, 128, 0, stream>>>(Ob, wot, bo, (float*)d_out);
}
